// MSARowAttentionWithPairBias_30975304139403
// MI455X (gfx1250) — matrix operand order repaired and checked by reading, not compiled, not run on hardware
//
#include <hip/hip_runtime.h>
#include <hip/hip_bf16.h>

// ---------------------------------------------------------------------------
// MSARowAttentionWithPairBias for MI455X (gfx1250), bf16 WMMA pipeline.
// Shapes: B=1, s=128, i=256, C_M=256, C_Z=128, h=8, c=32, hd=256.
// ---------------------------------------------------------------------------

#define S_DIM   128
#define I_DIM   256
#define CM      256
#define CZ      128
#define NH      8
#define CH      32
#define HD      256           // NH*CH
#define ROWS_M  (S_DIM * I_DIM)   // 32768
#define ROWS_Z  (I_DIM * I_DIM)   // 65536
#define PROW    264           // padded LDS row stride (bf16 elems) for P tiles

typedef __attribute__((ext_vector_type(16))) __bf16 v16bf;
typedef __attribute__((ext_vector_type(8)))  __bf16 v8bf;
typedef __attribute__((ext_vector_type(8)))  float  v8f;

union ABu { v16bf v; v8bf h[2]; };

static __device__ __forceinline__ v8f wmma_bf16(v16bf a, v16bf b, v8f c) {
    // D = A(16x32 bf16) * B(32x16 bf16) + C(16x16 f32)
    return __builtin_amdgcn_wmma_f32_16x16x32_bf16(
        false, a, false, b, (short)0, c, false, false);
}

// ---------------------------------------------------------------------------
// K0: f32 -> bf16 conversion (weights)
// ---------------------------------------------------------------------------
__global__ void cvt_bf16_kernel(const float* __restrict__ src,
                                __bf16* __restrict__ dst, int n) {
    int i = blockIdx.x * blockDim.x + threadIdx.x;
    if (i < n) dst[i] = (__bf16)src[i];
}

// ---------------------------------------------------------------------------
// K1: LayerNorm(m) over C_M=256; emit raw-bf16 and norm-bf16.
// One block (256 thr) per row.
// ---------------------------------------------------------------------------
__global__ void ln_m_kernel(const float* __restrict__ m,
                            const float* __restrict__ w,
                            const float* __restrict__ b,
                            __bf16* __restrict__ m_bf,
                            __bf16* __restrict__ mn_bf) {
    __shared__ float r1[CM];
    __shared__ float r2[CM];
    const int row = blockIdx.x;
    const int t   = threadIdx.x;
    const size_t idx = (size_t)row * CM + t;
    float x = m[idx];
    r1[t] = x; r2[t] = x * x;
    __syncthreads();
    for (int off = CM / 2; off > 0; off >>= 1) {
        if (t < off) { r1[t] += r1[t + off]; r2[t] += r2[t + off]; }
        __syncthreads();
    }
    const float mu  = r1[0] * (1.0f / CM);
    const float var = r2[0] * (1.0f / CM) - mu * mu;
    const float rstd = rsqrtf(var + 1e-5f);
    const float y = (x - mu) * rstd * w[t] + b[t];
    m_bf[idx]  = (__bf16)x;
    mn_bf[idx] = (__bf16)y;
}

// ---------------------------------------------------------------------------
// K2: LayerNorm(z) over C_Z=128 + pair bias: bias[h][i][j] = ln(z[i,j]) . Wb[h]
// One block (128 thr) per (i,j) pair.
// ---------------------------------------------------------------------------
__global__ void ln_z_bias_kernel(const float* __restrict__ z,
                                 const float* __restrict__ w,
                                 const float* __restrict__ b,
                                 const float* __restrict__ Wb,
                                 float* __restrict__ bias) {
    __shared__ float r1[CZ];
    __shared__ float r2[CZ];
    __shared__ float yr[CZ];
    const int p = blockIdx.x;      // i*256 + j
    const int t = threadIdx.x;
    float x = z[(size_t)p * CZ + t];
    r1[t] = x; r2[t] = x * x;
    __syncthreads();
    for (int off = CZ / 2; off > 0; off >>= 1) {
        if (t < off) { r1[t] += r1[t + off]; r2[t] += r2[t + off]; }
        __syncthreads();
    }
    const float mu  = r1[0] * (1.0f / CZ);
    const float var = r2[0] * (1.0f / CZ) - mu * mu;
    const float rstd = rsqrtf(var + 1e-5f);
    yr[t] = (x - mu) * rstd * w[t] + b[t];
    __syncthreads();
    for (int h = 0; h < NH; ++h) {
        r1[t] = yr[t] * Wb[h * CZ + t];
        __syncthreads();
        for (int off = CZ / 2; off > 0; off >>= 1) {
            if (t < off) r1[t] += r1[t + off];
            __syncthreads();
        }
        if (t == 0) bias[(size_t)h * ROWS_Z + p] = r1[0];
        __syncthreads();
    }
}

// ---------------------------------------------------------------------------
// K3: fused Q/K/V/gate projections.  One wave per 16x64 output tile
// (4 accumulators share one A operand -> 1 WMMA : 2.5 b128-loads).
//   which 0/1/2: q/k/v from m_norm; which 3: gate from raw m (sigmoid fused).
//   q,k stored [s,h,i,c]; v stored transposed [s,h,c,i]; gate [s,i,hd].
// ---------------------------------------------------------------------------
__global__ void qkvg_kernel(const __bf16* __restrict__ mn,
                            const __bf16* __restrict__ mraw,
                            const __bf16* __restrict__ Wq,
                            const __bf16* __restrict__ Wk,
                            const __bf16* __restrict__ Wv,
                            const __bf16* __restrict__ Wg,
                            const float*  __restrict__ bg,
                            __bf16* __restrict__ q,
                            __bf16* __restrict__ k,
                            __bf16* __restrict__ vT,
                            __bf16* __restrict__ gate) {
    const int wave  = (blockIdx.x * blockDim.x + threadIdx.x) >> 5;
    const int lane  = threadIdx.x & 31;
    const int which = wave & 3;                // output selector
    const int ng    = (wave >> 2) & 3;         // 64-col group
    const int mt    = wave >> 4;               // 0..2047 row tile
    const int ml    = lane & 15;
    const int half  = lane >> 4;

    const __bf16* src = (which == 3) ? mraw : mn;
    const __bf16* W   = (which == 0) ? Wq : (which == 1) ? Wk
                      : (which == 2) ? Wv : Wg;

    const __bf16* arow = src + (size_t)(mt * 16 + ml) * CM;
    const int col0 = ng * 64;
    const __bf16* bbase = W + (size_t)(col0 + ml) * CM + half * 16;

    __builtin_prefetch(arow, 0, 1);            // global_prefetch: warm A panel

    v8f acc[4] = {};
#pragma unroll
    for (int kt = 0; kt < 8; ++kt) {
        ABu a;
        a.h[0] = *(const v8bf*)(arow + kt * 32 + half * 16);
        a.h[1] = *(const v8bf*)(arow + kt * 32 + half * 16 + 8);
#pragma unroll
        for (int n = 0; n < 4; ++n) {
            v16bf bm = *(const v16bf*)(bbase + (size_t)n * 16 * CM + kt * 32);
            acc[n] = wmma_bf16(a.v, bm, acc[n]);
        }
    }

#pragma unroll
    for (int n = 0; n < 4; ++n) {
        const int col = col0 + n * 16 + ml;
#pragma unroll
        for (int r = 0; r < 8; ++r) {
            const int g  = mt * 16 + r + half * 8;   // global row in [0, 32768)
            const int s  = g >> 8;
            const int ii = g & 255;
            const float v = acc[n][r];
            if (which == 3) {
                const float gv = 1.0f / (1.0f + __expf(-(v + bg[col])));
                gate[(size_t)g * HD + col] = (__bf16)gv;
            } else {
                const int h = col >> 5, c = col & 31;
                if (which == 0)
                    q[(((size_t)s * NH + h) * I_DIM + ii) * CH + c] = (__bf16)v;
                else if (which == 1)
                    k[(((size_t)s * NH + h) * I_DIM + ii) * CH + c] = (__bf16)v;
                else
                    vT[(((size_t)s * NH + h) * CH + c) * I_DIM + ii] = (__bf16)v;
            }
        }
    }
}

// ---------------------------------------------------------------------------
// K4: attention per (s,h).  Block = 8 waves; each wave owns 2 row-tiles of 16.
//   scores(16x256) = Q Kt / sqrt(c) + bias + mask    (16 WMMAs, K=32=c)
//   softmax in registers (half-wave shfl_xor reductions), P -> LDS bf16
//   O(16x32) = P V                                   (16 WMMAs, K=256)
//   store gated O as bf16 [s,i,hd].
// ---------------------------------------------------------------------------
__global__ void attn_kernel(const __bf16* __restrict__ q,
                            const __bf16* __restrict__ k,
                            const __bf16* __restrict__ vT,
                            const float*  __restrict__ bias,
                            const float*  __restrict__ mask,
                            const __bf16* __restrict__ gate,
                            __bf16* __restrict__ og) {
    extern __shared__ __align__(16) char smem[];
    __bf16* pall = (__bf16*)smem;

    const int sh  = blockIdx.x;
    const int s   = sh >> 3;
    const int h   = sh & 7;
    const int wid  = threadIdx.x >> 5;
    const int lane = threadIdx.x & 31;
    const int ml   = lane & 15;
    const int half = lane >> 4;

    __bf16* pbuf = pall + (size_t)wid * 16 * PROW;

    const __bf16* qb = q  + ((size_t)s * NH + h) * I_DIM * CH;
    const __bf16* kb = k  + ((size_t)s * NH + h) * I_DIM * CH;
    const __bf16* vb = vT + ((size_t)s * NH + h) * CH * I_DIM;
    const float*  bb = bias + (size_t)h * ROWS_Z;
    const float*  mr = mask + (size_t)s * I_DIM;
    const float scale = 0.17677669529663687f;  // 1/sqrt(32)

    for (int rt = 0; rt < 2; ++rt) {
        const int mt = wid * 2 + rt;           // row tile 0..15

        // A operand: 16 rows of Q (shared across all 16 column tiles)
        ABu aq;
        const __bf16* arow = qb + (size_t)(mt * 16 + ml) * CH;
        aq.h[0] = *(const v8bf*)(arow + half * 16);
        aq.h[1] = *(const v8bf*)(arow + half * 16 + 8);

        v8f acc[16];
#pragma unroll
        for (int nt = 0; nt < 16; ++nt) {
            v16bf bm = *(const v16bf*)(kb + (size_t)(nt * 16 + ml) * CH + half * 16);
            v8f cz = {};
            acc[nt] = wmma_bf16(aq.v, bm, cz);
        }

        // scale + bias + mask, track row max (rows live in one 16-lane half)
        float mi[8], rmax[8], rsum[8];
#pragma unroll
        for (int r = 0; r < 8; ++r) {
            mi[r]   = mr[mt * 16 + r + half * 8];
            rmax[r] = -3.0e38f;
            rsum[r] = 0.0f;
        }
#pragma unroll
        for (int nt = 0; nt < 16; ++nt) {
            const int j = nt * 16 + ml;
            const float mj = mr[j];
#pragma unroll
            for (int r = 0; r < 8; ++r) {
                const int ii = mt * 16 + r + half * 8;
                float sc = (mi[r] > 0.0f && mj > 0.0f)
                         ? acc[nt][r] * scale + bb[(size_t)ii * I_DIM + j]
                         : -1.0e9f;
                acc[nt][r] = sc;
                rmax[r] = fmaxf(rmax[r], sc);
            }
        }
#pragma unroll
        for (int d = 1; d < 16; d <<= 1)
#pragma unroll
            for (int r = 0; r < 8; ++r)
                rmax[r] = fmaxf(rmax[r], __shfl_xor(rmax[r], d, 32));

#pragma unroll
        for (int nt = 0; nt < 16; ++nt)
#pragma unroll
            for (int r = 0; r < 8; ++r) {
                const float e = __expf(acc[nt][r] - rmax[r]);
                acc[nt][r] = e;
                rsum[r] += e;
            }
#pragma unroll
        for (int d = 1; d < 16; d <<= 1)
#pragma unroll
            for (int r = 0; r < 8; ++r)
                rsum[r] += __shfl_xor(rsum[r], d, 32);

        float rinv[8];
#pragma unroll
        for (int r = 0; r < 8; ++r) rinv[r] = 1.0f / rsum[r];

        // P (normalized, bf16) -> LDS in row-major A layout
#pragma unroll
        for (int nt = 0; nt < 16; ++nt) {
            const int j = nt * 16 + ml;
#pragma unroll
            for (int r = 0; r < 8; ++r)
                pbuf[(size_t)(r + half * 8) * PROW + j] =
                    (__bf16)(acc[nt][r] * rinv[r]);
        }

        // O = P * V   (two 16-col tiles over c, K = 256)
#pragma unroll
        for (int ct = 0; ct < 2; ++ct) {
            v8f ao = {};
#pragma unroll
            for (int kt = 0; kt < 8; ++kt) {
                ABu ap;
                const __bf16* prow = pbuf + (size_t)ml * PROW + kt * 32;
                ap.h[0] = *(const v8bf*)(prow + half * 16);
                ap.h[1] = *(const v8bf*)(prow + half * 16 + 8);
                v16bf bm = *(const v16bf*)(vb + (size_t)(ct * 16 + ml) * I_DIM
                                              + kt * 32 + half * 16);
                ao = wmma_bf16(ap.v, bm, ao);
            }
            const int c = ct * 16 + ml;
#pragma unroll
            for (int r = 0; r < 8; ++r) {
                const int ii = mt * 16 + r + half * 8;
                const size_t oidx = ((size_t)s * I_DIM + ii) * HD + h * CH + c;
                const float gv = (float)gate[oidx];
                og[oidx] = (__bf16)(ao[r] * gv);
            }
        }
    }
}

// ---------------------------------------------------------------------------
// K5: output projection: out = og @ Wo^T + bo   (f32 out)
// One wave per 16x64 tile (4 accumulators share one A operand).
// ---------------------------------------------------------------------------
__global__ void outproj_kernel(const __bf16* __restrict__ og,
                               const __bf16* __restrict__ Wo,
                               const float*  __restrict__ bo,
                               float* __restrict__ out) {
    const int wave = (blockIdx.x * blockDim.x + threadIdx.x) >> 5;
    const int lane = threadIdx.x & 31;
    const int ng   = wave & 3;                 // 64-col group
    const int mt   = wave >> 2;                // 0..2047
    const int ml   = lane & 15;
    const int half = lane >> 4;

    const __bf16* arow  = og + (size_t)(mt * 16 + ml) * HD;
    const int col0 = ng * 64;
    const __bf16* bbase = Wo + (size_t)(col0 + ml) * HD + half * 16;

    __builtin_prefetch(arow, 0, 1);

    v8f acc[4] = {};
#pragma unroll
    for (int kt = 0; kt < 8; ++kt) {
        ABu a;
        a.h[0] = *(const v8bf*)(arow + kt * 32 + half * 16);
        a.h[1] = *(const v8bf*)(arow + kt * 32 + half * 16 + 8);
#pragma unroll
        for (int n = 0; n < 4; ++n) {
            v16bf bm = *(const v16bf*)(bbase + (size_t)n * 16 * HD + kt * 32);
            acc[n] = wmma_bf16(a.v, bm, acc[n]);
        }
    }

#pragma unroll
    for (int n = 0; n < 4; ++n) {
        const int col = col0 + n * 16 + ml;
        const float bn = bo[col];
#pragma unroll
        for (int r = 0; r < 8; ++r) {
            const int g = mt * 16 + r + half * 8;
            out[(size_t)g * CM + col] = acc[n][r] + bn;
        }
    }
}

// ---------------------------------------------------------------------------
// Host-side launch
// ---------------------------------------------------------------------------
extern "C" void kernel_launch(void* const* d_in, const int* in_sizes, int n_in,
                              void* d_out, int out_size, void* d_ws, size_t ws_size,
                              hipStream_t stream) {
    const float* m_in   = (const float*)d_in[0];
    const float* z_in   = (const float*)d_in[1];
    const float* msk    = (const float*)d_in[2];
    const float* lnm_w  = (const float*)d_in[3];
    const float* lnm_b  = (const float*)d_in[4];
    const float* lnz_w  = (const float*)d_in[5];
    const float* lnz_b  = (const float*)d_in[6];
    const float* Wq     = (const float*)d_in[7];
    const float* Wk     = (const float*)d_in[8];
    const float* Wv     = (const float*)d_in[9];
    const float* Wb     = (const float*)d_in[10];
    const float* Wg     = (const float*)d_in[11];
    const float* bg     = (const float*)d_in[12];
    const float* Wo     = (const float*)d_in[13];
    const float* bo     = (const float*)d_in[14];
    float* out = (float*)d_out;

    // scratch carve-out (256B aligned chunks)
    char* p = (char*)d_ws;
    size_t off = 0;
    auto alloc = [&](size_t bytes) -> void* {
        void* r = p + off;
        off += (bytes + 255) & ~(size_t)255;
        return r;
    };
    const size_t wbytes  = (size_t)CM * CM * sizeof(__bf16);       // 128 KB
    const size_t actB    = (size_t)ROWS_M * CM * sizeof(__bf16);   // 16 MB
    __bf16* wq_bf = (__bf16*)alloc(wbytes);
    __bf16* wk_bf = (__bf16*)alloc(wbytes);
    __bf16* wv_bf = (__bf16*)alloc(wbytes);
    __bf16* wg_bf = (__bf16*)alloc(wbytes);
    __bf16* wo_bf = (__bf16*)alloc(wbytes);
    __bf16* m_bf  = (__bf16*)alloc(actB);
    __bf16* mn_bf = (__bf16*)alloc(actB);
    __bf16* q_bf  = (__bf16*)alloc(actB);
    __bf16* k_bf  = (__bf16*)alloc(actB);
    __bf16* vT_bf = (__bf16*)alloc(actB);
    __bf16* gt_bf = (__bf16*)alloc(actB);
    __bf16* og_bf = (__bf16*)alloc(actB);
    float*  bias  = (float*)alloc((size_t)NH * ROWS_Z * sizeof(float)); // 2 MB

    const int wn = CM * CM;
    cvt_bf16_kernel<<<dim3((wn + 255) / 256), dim3(256), 0, stream>>>(Wq, wq_bf, wn);
    cvt_bf16_kernel<<<dim3((wn + 255) / 256), dim3(256), 0, stream>>>(Wk, wk_bf, wn);
    cvt_bf16_kernel<<<dim3((wn + 255) / 256), dim3(256), 0, stream>>>(Wv, wv_bf, wn);
    cvt_bf16_kernel<<<dim3((wn + 255) / 256), dim3(256), 0, stream>>>(Wg, wg_bf, wn);
    cvt_bf16_kernel<<<dim3((wn + 255) / 256), dim3(256), 0, stream>>>(Wo, wo_bf, wn);

    ln_m_kernel<<<dim3(ROWS_M), dim3(CM), 0, stream>>>(m_in, lnm_w, lnm_b, m_bf, mn_bf);
    ln_z_bias_kernel<<<dim3(ROWS_Z), dim3(CZ), 0, stream>>>(z_in, lnz_w, lnz_b, Wb, bias);

    // 2048 row-tiles * 4 col-groups * 4 outputs = 32768 waves, 8 waves/block
    qkvg_kernel<<<dim3(32768 / 8), dim3(256), 0, stream>>>(
        mn_bf, m_bf, wq_bf, wk_bf, wv_bf, wg_bf, bg, q_bf, k_bf, vT_bf, gt_bf);

    // one block per (s,h); dynamic LDS: 8 waves * 16 rows * PROW bf16
    const size_t attn_lds = (size_t)8 * 16 * PROW * sizeof(__bf16); // 67584 B
    attn_kernel<<<dim3(S_DIM * NH), dim3(256), attn_lds, stream>>>(
        q_bf, k_bf, vT_bf, bias, msk, gt_bf, og_bf);

    // 2048 row-tiles * 4 col-groups = 8192 waves, 8 waves/block
    outproj_kernel<<<dim3(8192 / 8), dim3(256), 0, stream>>>(og_bf, wo_bf, bo, out);
}
